// ARSLMModel_23244363006348
// MI455X (gfx1250) — matrix operand order repaired and checked by reading, not compiled, not run on hardware
//
#include <hip/hip_runtime.h>
#include <hip/hip_bf16.h>
#include <math.h>

typedef __attribute__((ext_vector_type(16))) _Float16 v16h;
typedef __attribute__((ext_vector_type(8)))  _Float16 v8h;
typedef __attribute__((ext_vector_type(8)))  float    v8f;

#define WMMA_F32_F16(a, b, c) \
  __builtin_amdgcn_wmma_f32_16x16x32_f16(false, (a), false, (b), (short)0, (c), false, false)

constexpr int Vv  = 32000;
constexpr int Ee  = 64;
constexpr int Hh  = 64;
constexpr int Ll  = 2;
constexpr int Bb  = 32;
constexpr int Ss  = 256;
constexpr int CTX = 192;   // 2H + E
constexpr int NT  = Vv / 16;   // 2000 N tiles in the head

// ---- WMMA operand loaders (16x16x32 f16; contiguous k order, same for A and B) ----
// A (16x32, MxK): lane half h holds K {kb+16h .. kb+16h+15}
__device__ inline v16h load_a_f16(const _Float16* base, int ldm, int row, int kbase, int half) {
  const _Float16* p = base + row * ldm + kbase + 16 * half;
  v8h lo = *(const v8h*)(p);
  v8h hi = *(const v8h*)(p + 8);
  v16h a;
#pragma unroll
  for (int e = 0; e < 8; ++e) { a[e] = lo[e]; a[e + 8] = hi[e]; }
  return a;
}

// B (32x16, KxN) from row-major [K][N]: lane half h holds K {kb+16h .. kb+16h+15} at column ncol
__device__ inline v16h load_b_f16(const _Float16* W, int ldw, int kbase, int ncol, int half) {
  v16h b;
#pragma unroll
  for (int e = 0; e < 16; ++e)
    b[e] = W[(size_t)(kbase + 16 * half + e) * ldw + ncol];
  return b;
}

// Packed B fragment: 16 contiguous f16 per lane -> two global_load_b128
__device__ inline v16h load_b_packed(const _Float16* p) {
  v8h lo = *(const v8h*)(p);
  v8h hi = *(const v8h*)(p + 8);
  v16h b;
#pragma unroll
  for (int e = 0; e < 8; ++e) { b[e] = lo[e]; b[e + 8] = hi[e]; }
  return b;
}

// ---- fp32 -> fp16 conversion ----
__global__ void cvt_f32_f16(const float* __restrict__ src, _Float16* __restrict__ dst, int n) {
  int i = blockIdx.x * blockDim.x + threadIdx.x;
  int stride = gridDim.x * blockDim.x;
  for (; i < n; i += stride) dst[i] = (_Float16)src[i];
}

// ---- Pre-shuffle headW [64][32000] f32 into per-lane WMMA-B layout f16 ----
// Layout: frag c = nt*2 + kc; element [ (c*32 + lane)*16 + e ] =
//         W[ kc*32 + 16*(lane>>4) + e ][ nt*16 + (lane&15) ]
// One thread per (c, lane): 500 blocks * 256 threads = 128000 = 2000*2*32.
__global__ void pack_headW(const float* __restrict__ W, _Float16* __restrict__ dst) {
  int idx  = blockIdx.x * blockDim.x + threadIdx.x;   // 0 .. 127999
  int ln   = idx & 31;
  int c    = idx >> 5;          // 0 .. 3999
  int kc   = c & 1;
  int nt   = c >> 1;
  int ncol = nt * 16 + (ln & 15);
  int krow = kc * 32 + 16 * (ln >> 4);
  _Float16* d = dst + (size_t)idx * 16;
#pragma unroll
  for (int e = 0; e < 16; ++e)
    d[e] = (_Float16)W[(size_t)(krow + e) * Vv + ncol];
}

// ---- Persistent recurrent scan: 1 block, 8 wave32s, all state in LDS ----
__global__ void __launch_bounds__(256) scan_kernel(
    const int* __restrict__ ids, const float* __restrict__ emb,
    const _Float16* __restrict__ W1h, const float* __restrict__ b1,
    const _Float16* __restrict__ W2h, const float* __restrict__ b2,
    const float* __restrict__ Wg, const float* __restrict__ bg,
    const float* __restrict__ lng, const float* __restrict__ lnb,
    _Float16* __restrict__ hseq)   // [B*S, H] f16, row m = b*S + t
{
  __shared__ __align__(16) _Float16 hF16[Ll][2][Bb][Hh];  // ping-pong f16 h states (h1/h2)
  __shared__ __align__(16) float    h1F32[Ll][Bb][Hh];    // f32 current h1 (residual path)
  __shared__ __align__(16) float    xF32[Bb][Ee];
  __shared__ __align__(16) _Float16 xF16[Bb][Ee];
  __shared__ __align__(16) _Float16 aF16[Bb][Hh];         // relu activations
  __shared__ __align__(16) float    hpre[Bb][Hh];         // pre-LN h
  __shared__ float gate_s[Bb];
  __shared__ float gpart[Bb][8];

  const int tid  = threadIdx.x;
  const int lane = tid & 31;
  const int wv   = tid >> 5;          // wave 0..7
  const int half = (lane >> 4) & 1;
  const int r    = lane & 15;
  const int mt   = wv & 1;            // M tile (batch rows 0-15 / 16-31)
  const int nt   = wv >> 1;           // N tile (hidden cols, 4 tiles)

  // Preload this wave's time-invariant B tiles of W1/W2 into registers.
  v16h bW1[Ll][6];
  v16h bW2[Ll][2];
#pragma unroll
  for (int l = 0; l < Ll; ++l) {
#pragma unroll
    for (int kc = 0; kc < 6; ++kc)
      bW1[l][kc] = load_b_f16(W1h + (size_t)l * CTX * Hh, Hh, kc * 32, nt * 16 + r, half);
#pragma unroll
    for (int kc = 0; kc < 2; ++kc)
      bW2[l][kc] = load_b_f16(W2h + (size_t)l * Hh * Hh, Hh, kc * 32, nt * 16 + r, half);
  }

  // zero initial hidden state
  for (int i = tid; i < Ll * 2 * Bb * Hh; i += 256) ((_Float16*)hF16)[i] = (_Float16)0.f;
  for (int i = tid; i < Ll * Bb * Hh; i += 256)     ((float*)h1F32)[i] = 0.f;
  __syncthreads();

  for (int t = 0; t < Ss; ++t) {
    const int p = t & 1;  // h1 = hF16[l][p], h2 = hF16[l][p^1]; LN writes into [p^1]

    // ---- embedding gather into xF32 / xF16 ----
    for (int i = tid; i < Bb * Ee; i += 256) {
      int row = i >> 6, col = i & 63;
      int id = ids[row * Ss + t];
      float v = emb[(size_t)id * Ee + col];
      xF32[row][col] = v;
      xF16[row][col] = (_Float16)v;
    }
    __syncthreads();

#pragma unroll
    for (int l = 0; l < Ll; ++l) {
      const _Float16* h1h = &hF16[l][p][0][0];
      const _Float16* h2h = &hF16[l][p ^ 1][0][0];
      const _Float16* xh  = (l == 0) ? &xF16[0][0] : &hF16[0][p ^ 1][0][0];   // layer1 x = layer0 new h
      const float*    xf  = (l == 0) ? &xF32[0][0] : &h1F32[0][0][0];

      // ---- gate partial dot c @ Wg (all 256 threads, 24 MACs each) ----
      {
        int row = tid >> 3, seg = tid & 7;
        float s = 0.f;
        for (int k = seg * 24; k < seg * 24 + 24; ++k) {
          float cv = (k < 64)  ? (float)h1h[row * 64 + k]
                   : (k < 128) ? (float)h2h[row * 64 + (k - 64)]
                               : (float)xh[row * 64 + (k - 128)];
          s += cv * Wg[l * CTX + k];
        }
        gpart[row][seg] = s;
      }

      // ---- GEMM1: [32,192] @ [192,64], one 16x16 tile per wave, K = 6 chunks ----
      v8f acc = {0.f, 0.f, 0.f, 0.f, 0.f, 0.f, 0.f, 0.f};
#pragma unroll
      for (int kc = 0; kc < 6; ++kc) {
        const _Float16* src = (kc < 2) ? h1h : (kc < 4) ? h2h : xh;
        v16h a = load_a_f16(src, 64, mt * 16 + r, (kc & 1) * 32, half);
        acc = WMMA_F32_F16(a, bW1[l][kc], acc);
      }
      {
        int nc = nt * 16 + r;
        float bb = b1[l * Hh + nc];
#pragma unroll
        for (int i = 0; i < 8; ++i) {
          float v = acc[i] + bb;
          aF16[mt * 16 + i + 8 * half][nc] = (_Float16)(v > 0.f ? v : 0.f);  // relu
        }
      }
      __syncthreads();

      // ---- GEMM2: [32,64] @ [64,64]; wave0 also reduces the gate ----
      v8f acc2 = {0.f, 0.f, 0.f, 0.f, 0.f, 0.f, 0.f, 0.f};
#pragma unroll
      for (int kc = 0; kc < 2; ++kc) {
        v16h a = load_a_f16(&aF16[0][0], 64, mt * 16 + r, kc * 32, half);
        acc2 = WMMA_F32_F16(a, bW2[l][kc], acc2);
      }
      if (tid < Bb) {
        float s = bg[l];
#pragma unroll
        for (int j = 0; j < 8; ++j) s += gpart[tid][j];
        gate_s[tid] = 1.f / (1.f + __expf(-s));
      }
      __syncthreads();

      // ---- residual: h = h1 + gate*cand + 0.1*x ----
      {
        int nc = nt * 16 + r;
        float b2v = b2[l * Hh + nc];
#pragma unroll
        for (int i = 0; i < 8; ++i) {
          int row = mt * 16 + i + 8 * half;
          float cand = acc2[i] + b2v;
          hpre[row][nc] = h1F32[l][row][nc] + gate_s[row] * cand + 0.1f * xf[row * 64 + nc];
        }
      }
      __syncthreads();

      // ---- LayerNorm (one row per thread, threads 0..31) + state rotation ----
      if (tid < Bb) {
        int row = tid;
        float s = 0.f, ss = 0.f;
        for (int c = 0; c < Hh; ++c) { float v = hpre[row][c]; s += v; ss += v * v; }
        float mu  = s * (1.f / Hh);
        float var = ss * (1.f / Hh) - mu * mu;
        float rs  = rsqrtf(var + 1e-5f);
        for (int c = 0; c < Hh; ++c) {
          float y = (hpre[row][c] - mu) * rs * lng[l * Hh + c] + lnb[l * Hh + c];
          h1F32[l][row][c] = y;
          hF16[l][p ^ 1][row][c] = (_Float16)y;        // becomes h1 next step; old h1 becomes h2
          if (l == Ll - 1) hseq[((size_t)row * Ss + t) * Hh + c] = (_Float16)y;
        }
      }
      __syncthreads();
    }
  }
}

// ---- Head: [8192,64] @ [64,32000] + bias, WMMA, store-bandwidth bound ----
// 1 wave = one 16-row M tile x 4 consecutive 16-col N tiles (A regs reused).
// B fragments pre-packed per lane -> two b128 loads each. NT stores for the
// 1 GB logits stream so L2 keeps headW/hseq resident.
__global__ void __launch_bounds__(256) head_kernel(
    const _Float16* __restrict__ hseq, const _Float16* __restrict__ Wpk,
    const float* __restrict__ headb, float* __restrict__ out)
{
  const int tid  = threadIdx.x;
  const int lane = tid & 31;
  const int wv   = tid >> 5;
  const int half = (lane >> 4) & 1;
  const int r    = lane & 15;

  int g  = blockIdx.x * 8 + wv;   // 0..255999
  int mt = g / 500;               // 0..511
  int ng = g % 500;               // 0..499
  int row = mt * 16 + r;

  v16h a0 = load_a_f16(hseq, Hh, row, 0, half);
  v16h a1 = load_a_f16(hseq, Hh, row, 32, half);

#pragma unroll
  for (int j = 0; j < 4; ++j) {
    int ntile = ng * 4 + j;
    int ncol  = ntile * 16 + r;
    const _Float16* bp = Wpk + ((size_t)(ntile * 2) * 32 + lane) * 16;
    v16h b0 = load_b_packed(bp);            // kc = 0 fragment
    v16h b1 = load_b_packed(bp + 32 * 16);  // kc = 1 fragment
    v8f acc = {0.f, 0.f, 0.f, 0.f, 0.f, 0.f, 0.f, 0.f};
    acc = WMMA_F32_F16(a0, b0, acc);
    acc = WMMA_F32_F16(a1, b1, acc);
    float bias = headb[ncol];
#pragma unroll
    for (int i = 0; i < 8; ++i)
      __builtin_nontemporal_store(acc[i] + bias,
          &out[(size_t)(mt * 16 + i + 8 * half) * Vv + ncol]);  // 64B-coalesced rows, NT
  }
}

extern "C" void kernel_launch(void* const* d_in, const int* in_sizes, int n_in,
                              void* d_out, int out_size, void* d_ws, size_t ws_size,
                              hipStream_t stream) {
  const int*   ids   = (const int*)d_in[0];
  const float* emb   = (const float*)d_in[1];
  const float* W1    = (const float*)d_in[2];
  const float* b1    = (const float*)d_in[3];
  const float* W2    = (const float*)d_in[4];
  const float* b2    = (const float*)d_in[5];
  const float* Wg    = (const float*)d_in[6];
  const float* bg    = (const float*)d_in[7];
  const float* lng   = (const float*)d_in[8];
  const float* lnb   = (const float*)d_in[9];
  const float* headW = (const float*)d_in[10];
  const float* headb = (const float*)d_in[11];
  float* out = (float*)d_out;

  // workspace layout (bytes): packed headW f16 | hseq f16 | W1 f16 | W2 f16
  char* ws = (char*)d_ws;
  _Float16* headWp = (_Float16*)(ws);                               // 64*32000  -> 4,096,000 B
  _Float16* hseq   = (_Float16*)(ws + 4096000);                     // 8192*64   -> 1,048,576 B
  _Float16* W1h    = (_Float16*)(ws + 4096000 + 1048576);           // 2*192*64  ->    49,152 B
  _Float16* W2h    = (_Float16*)(ws + 4096000 + 1048576 + 49152);   // 2*64*64   ->    16,384 B

  pack_headW <<<500, 256, 0, stream>>>(headW, headWp);              // 2000*2*32 lanes * 16 f16
  cvt_f32_f16<<<96,  256, 0, stream>>>(W1, W1h, Ll * CTX * Hh);
  cvt_f32_f16<<<32,  256, 0, stream>>>(W2, W2h, Ll * Hh * Hh);

  scan_kernel<<<1, 256, 0, stream>>>(ids, emb, W1h, b1, W2h, b2, Wg, bg, lng, lnb, hseq);

  head_kernel<<<32000, 256, 0, stream>>>(hseq, headWp, headb, out);
}
